// PatchwiseStructureTensorLoss_2035814499040
// MI455X (gfx1250) — matrix operand order repaired and checked by reading, not compiled, not run on hardware
//
#include <hip/hip_runtime.h>
#include <math.h>

// ---------------- types for WMMA ----------------
typedef __attribute__((ext_vector_type(16))) _Float16 v16h;
typedef __attribute__((ext_vector_type(8)))  _Float16 v8h;
typedef __attribute__((ext_vector_type(8)))  float    v8f;

// ---------------- problem constants ----------------
#define BB   4
#define HH   192
#define WW   192
#define NQ   4096          // 64*64 patches at full res
#define NM   5376          // 4096 + 1024 + 256
#define FEAT 27
#define KPAD 32

// Gaussian kernels (sigma=1, rho=10), radius-1
#define GS0 0.27406862f
#define GS1 0.45186276f
#define GR0 0.33277750f
#define GR1 0.33444500f

// ---------------- bicubic resize (a=-0.5, frac=0.5 taps) ----------------
__global__ void pst_resize_kernel(const float* __restrict__ in, float* __restrict__ out,
                                  int Hin, int Win, int Hout, int Wout, int scale, int start) {
    int t = blockIdx.x * blockDim.x + threadIdx.x;
    int total = BB * 3 * Hout * Wout;
    if (t >= total) return;
    int hw = Hout * Wout;
    int bc = t / hw;
    int r  = t - bc * hw;
    int oy = r / Wout, ox = r - oy * Wout;
    const float w4[4] = {-0.0625f, 0.5625f, 0.5625f, -0.0625f};
    const float* src = in + (size_t)bc * Hin * Win;
    float s = 0.f;
    #pragma unroll
    for (int i = 0; i < 4; ++i) {
        int yy = oy * scale + start + i;
        yy = yy < 0 ? 0 : (yy > Hin - 1 ? Hin - 1 : yy);
        float rowacc = 0.f;
        #pragma unroll
        for (int j = 0; j < 4; ++j) {
            int xx = ox * scale + start + j;
            xx = xx < 0 ? 0 : (xx > Win - 1 ? Win - 1 : xx);
            rowacc += w4[j] * src[(size_t)yy * Win + xx];
        }
        s += w4[i] * rowacc;
    }
    out[t] = s;
}

// ---------------- 3x3 zero-padded conv on a 3x3 patch ----------------
__device__ __forceinline__ void conv3z(const float in[3][3], const float k[3][3], float out[3][3]) {
    #pragma unroll
    for (int y = 0; y < 3; ++y)
    #pragma unroll
    for (int x = 0; x < 3; ++x) {
        float s = 0.f;
        #pragma unroll
        for (int i = 0; i < 3; ++i) {
            int yy = y + i - 1;
            if (yy < 0 || yy > 2) continue;
            #pragma unroll
            for (int j = 0; j < 3; ++j) {
                int xx = x + j - 1;
                if (xx < 0 || xx > 2) continue;
                s += k[i][j] * in[yy][xx];
            }
        }
        out[y][x] = s;
    }
}

// ---------------- per-patch structure tensor features ----------------
__global__ void pst_patches_kernel(const float* __restrict__ img, float* __restrict__ out,
                                   int H, int W, int ldRows, int rowOff) {
    int Hb = H / 3, Wb = W / 3, P = Hb * Wb;
    int t = blockIdx.x * blockDim.x + threadIdx.x;
    if (t >= BB * P) return;
    int b = t / P, p = t - b * P;
    int py = p / Wb, px = p - py * Wb;
    size_t plane = (size_t)H * W;
    const float* base = img + (size_t)b * 3 * plane + (size_t)(py * 3) * W + px * 3;

    float gray[3][3];
    #pragma unroll
    for (int y = 0; y < 3; ++y)
    #pragma unroll
    for (int x = 0; x < 3; ++x)
        gray[y][x] = 0.2989f * base[(size_t)y * W + x]
                   + 0.587f  * base[plane + (size_t)y * W + x]
                   + 0.114f  * base[2 * plane + (size_t)y * W + x];

    const float gs[3]  = {GS0, GS1, GS0};
    const float dgs[3] = {GS0, 0.f, -GS0};
    const float gr[3]  = {GR0, GR1, GR0};
    float KX[3][3], KY[3][3], KS[3][3];
    #pragma unroll
    for (int i = 0; i < 3; ++i)
    #pragma unroll
    for (int j = 0; j < 3; ++j) {
        KX[i][j] = gs[i] * dgs[j];
        KY[i][j] = dgs[i] * gs[j];
        KS[i][j] = gr[i] * gr[j];
    }
    float Ix[3][3], Iy[3][3];
    conv3z(gray, KX, Ix);
    conv3z(gray, KY, Iy);
    float Pxx[3][3], Pyy[3][3], Pxy[3][3];
    #pragma unroll
    for (int y = 0; y < 3; ++y)
    #pragma unroll
    for (int x = 0; x < 3; ++x) {
        Pxx[y][x] = Ix[y][x] * Ix[y][x];
        Pyy[y][x] = Iy[y][x] * Iy[y][x];
        Pxy[y][x] = Ix[y][x] * Iy[y][x];
    }
    float Jxx[3][3], Jyy[3][3], Jxy[3][3];
    conv3z(Pxx, KS, Jxx);
    conv3z(Pyy, KS, Jyy);
    conv3z(Pxy, KS, Jxy);

    float J[FEAT];
    #pragma unroll
    for (int y = 0; y < 3; ++y)
    #pragma unroll
    for (int x = 0; x < 3; ++x) {
        J[y * 3 + x]      = Jxx[y][x];
        J[9 + y * 3 + x]  = Jyy[y][x];
        J[18 + y * 3 + x] = Jxy[y][x];
    }
    float n = 0.f;
    #pragma unroll
    for (int k = 0; k < FEAT; ++k) n += J[k] * J[k];
    float inv = 1.0f / (sqrtf(n) + 1e-12f);
    float* dst = out + ((size_t)b * ldRows + rowOff + p) * FEAT;
    #pragma unroll
    for (int k = 0; k < FEAT; ++k) dst[k] = J[k] * inv;
}

// ---------------- pack f32 features -> f16 GEMM operands (K padded to 32) ----
__global__ void pst_pack_kernel(const float* __restrict__ p1, const float* __restrict__ p2c,
                                _Float16* __restrict__ q16, _Float16* __restrict__ b16) {
    int t = blockIdx.x * blockDim.x + threadIdx.x;
    if (t >= BB * NM) return;
    int b = t / NM, m = t - b * NM;
    const float* src = p2c + ((size_t)b * NM + m) * FEAT;
    _Float16* db = b16 + ((size_t)b * NM + m) * KPAD;
    #pragma unroll
    for (int k = 0; k < KPAD; ++k) db[k] = (k < FEAT) ? (_Float16)src[k] : (_Float16)0.f;
    if (m < NQ) {   // first NQ rows of p2c are p2 (same patch grid as p1): q = p1 + p2
        const float* s1 = p1 + ((size_t)b * NQ + m) * FEAT;
        _Float16* dq = q16 + ((size_t)b * NQ + m) * KPAD;
        #pragma unroll
        for (int k = 0; k < KPAD; ++k) dq[k] = (k < FEAT) ? (_Float16)(s1[k] + src[k]) : (_Float16)0.f;
    }
}

// ---------------- helpers for the WMMA argmax kernel ----------------
__device__ __forceinline__ v16h pst_load_a_frag(const _Float16* qrow, int half_sel) {
    // A fragment 16x32 f16, contiguous K: lane<16 holds K{0-15}, lane>=16 K{16-31}
    v8h lo = *(const v8h*)(qrow + half_sel * 16);
    v8h hi = *(const v8h*)(qrow + half_sel * 16 + 8);
    v16h a;
    #pragma unroll
    for (int i = 0; i < 8; ++i) { a[i] = lo[i]; a[8 + i] = hi[i]; }
    return a;
}

__device__ __forceinline__ void pst_update_best(const v8f& c, int col,
                                                float* bestv, int* besti) {
    #pragma unroll
    for (int r = 0; r < 8; ++r) {
        float v = c[r];
        if (v > bestv[r]) { bestv[r] = v; besti[r] = col; }   // strict > keeps earliest m
    }
}

__device__ __forceinline__ void pst_reduce_write(float* bestv, int* besti,
                                                 int lsub, int half_sel,
                                                 int* ind, size_t rowbase) {
    #pragma unroll
    for (int r = 0; r < 8; ++r) {
        float v = bestv[r];
        int   idx = besti[r];
        #pragma unroll
        for (int m = 1; m < 16; m <<= 1) {     // xor<16 stays inside each 16-lane half
            float ov = __shfl_xor(v, m, 32);
            int   oi = __shfl_xor(idx, m, 32);
            if (ov > v || (ov == v && oi < idx)) { v = ov; idx = oi; }
        }
        if (lsub == 0) {
            int row = r + 8 * half_sel;
            ind[rowbase + row] = idx;
        }
    }
}

// ---------------- WMMA nearest-neighbor (argmax of q . b^T) ----------------
// 1 wave / 32-row tile of q (two A fragments). Column loop unrolled by 2 B tiles:
// 4 independent v_wmma_f32_16x16x32_f16 per iteration, each B load feeds 2 WMMAs.
__global__ void __launch_bounds__(32) pst_argmax_wmma_kernel(
        const _Float16* __restrict__ q16, const _Float16* __restrict__ b16,
        int* __restrict__ ind) {
    const int ntiles = NQ >> 5;               // 32 rows per wave
    int wg   = blockIdx.x;
    int b    = wg / ntiles;
    int tile = wg - b * ntiles;
    int lane = threadIdx.x;
    int half_sel = lane >> 4;
    int lsub = lane & 15;

    const _Float16* qrow0 = q16 + ((size_t)b * NQ + (size_t)tile * 32 + lsub) * KPAD;
    v16h a0 = pst_load_a_frag(qrow0, half_sel);
    v16h a1 = pst_load_a_frag(qrow0 + 16 * KPAD, half_sel);

    float bestv0[8], bestv1[8];
    int   besti0[8], besti1[8];
    #pragma unroll
    for (int r = 0; r < 8; ++r) {
        bestv0[r] = -1e30f; besti0[r] = 0;
        bestv1[r] = -1e30f; besti1[r] = 0;
    }

    const _Float16* bbase = b16 + (size_t)b * NM * KPAD;
    for (int mt = 0; mt < (NM >> 4); mt += 2) {
        // B fragment 32x16 f16: col N = lsub, element e <-> K = half_sel*16 + e (contiguous 32B)
        const _Float16* brow0 = bbase + ((size_t)mt * 16 + lsub) * KPAD + half_sel * 16;
        const _Float16* brow1 = brow0 + 16 * KPAD;
        __builtin_prefetch(brow0 + 32 * KPAD, 0, 1);   // next pair of tiles
        __builtin_prefetch(brow1 + 32 * KPAD, 0, 1);
        v16h bf0 = *(const v16h*)brow0;
        v16h bf1 = *(const v16h*)brow1;

        v8f c00 = {}, c10 = {}, c01 = {}, c11 = {};
        c00 = __builtin_amdgcn_wmma_f32_16x16x32_f16(false, a0, false, bf0, (short)0, c00, false, false);
        c10 = __builtin_amdgcn_wmma_f32_16x16x32_f16(false, a1, false, bf0, (short)0, c10, false, false);
        c01 = __builtin_amdgcn_wmma_f32_16x16x32_f16(false, a0, false, bf1, (short)0, c01, false, false);
        c11 = __builtin_amdgcn_wmma_f32_16x16x32_f16(false, a1, false, bf1, (short)0, c11, false, false);

        int col0 = mt * 16 + lsub;
        int col1 = col0 + 16;
        pst_update_best(c00, col0, bestv0, besti0);
        pst_update_best(c10, col0, bestv1, besti1);
        pst_update_best(c01, col1, bestv0, besti0);
        pst_update_best(c11, col1, bestv1, besti1);
    }

    size_t rowbase = (size_t)b * NQ + (size_t)tile * 32;
    pst_reduce_write(bestv0, besti0, lsub, half_sel, ind, rowbase);
    pst_reduce_write(bestv1, besti1, lsub, half_sel, ind, rowbase + 16);
}

// ---------------- gather nearest neighbor + L1 accumulate ----------------
__global__ void pst_zero_kernel(float* acc) { if (threadIdx.x == 0 && blockIdx.x == 0) acc[0] = 0.f; }

__global__ void pst_gather_l1_kernel(const float* __restrict__ p1, const float* __restrict__ p2c,
                                     const int* __restrict__ ind, float* __restrict__ acc) {
    int t = blockIdx.x * blockDim.x + threadIdx.x;
    float s = 0.f;
    if (t < BB * NQ) {
        int b = t / NQ;
        const float* pr = p1 + (size_t)t * FEAT;
        const float* sr = p2c + ((size_t)b * NM + ind[t]) * FEAT;
        #pragma unroll
        for (int k = 0; k < FEAT; ++k) s += fabsf(pr[k] - sr[k]);
    }
    #pragma unroll
    for (int m = 16; m >= 1; m >>= 1) s += __shfl_down(s, m, 32);
    if ((threadIdx.x & 31) == 0) atomicAdd(acc, s);
}

__global__ void pst_finalize_kernel(const float* acc, float* out) {
    if (threadIdx.x == 0 && blockIdx.x == 0)
        out[0] = acc[0] * (1.0f / ((float)BB * NQ * FEAT));
}

// ---------------- launcher ----------------
extern "C" void kernel_launch(void* const* d_in, const int* in_sizes, int n_in,
                              void* d_out, int out_size, void* d_ws, size_t ws_size,
                              hipStream_t stream) {
    (void)in_sizes; (void)n_in; (void)out_size; (void)ws_size;
    const float* x  = (const float*)d_in[0];
    const float* gt = (const float*)d_in[1];
    float* out = (float*)d_out;

    char* ws = (char*)d_ws;
    size_t off = 0;
    auto carve = [&](size_t bytes) { size_t r = off; off = (off + bytes + 255) & ~(size_t)255; return r; };
    float*    gt_half = (float*)   (ws + carve((size_t)BB * 3 * 96 * 96 * 4));
    float*    gt_qtr  = (float*)   (ws + carve((size_t)BB * 3 * 48 * 48 * 4));
    float*    p1      = (float*)   (ws + carve((size_t)BB * NQ * FEAT * 4));
    float*    p2c     = (float*)   (ws + carve((size_t)BB * NM * FEAT * 4));
    _Float16* q16     = (_Float16*)(ws + carve((size_t)BB * NQ * KPAD * 2));
    _Float16* b16     = (_Float16*)(ws + carve((size_t)BB * NM * KPAD * 2));
    int*      ind     = (int*)     (ws + carve((size_t)BB * NQ * 4));
    float*    acc     = (float*)   (ws + carve(4));

    const int T = 256;
    // 1) resizes of gt (bicubic, fixed 4-tap filters)
    int nh = BB * 3 * 96 * 96;
    pst_resize_kernel<<<(nh + T - 1) / T, T, 0, stream>>>(gt, gt_half, HH, WW, 96, 96, 2, -1);
    int nq4 = BB * 3 * 48 * 48;
    pst_resize_kernel<<<(nq4 + T - 1) / T, T, 0, stream>>>(gt, gt_qtr, HH, WW, 48, 48, 4, 0);

    // 2) structure-tensor patch features
    pst_patches_kernel<<<(BB * NQ + T - 1) / T, T, 0, stream>>>(x,  p1,  HH, WW, NQ, 0);
    pst_patches_kernel<<<(BB * NQ + T - 1) / T, T, 0, stream>>>(gt, p2c, HH, WW, NM, 0);
    pst_patches_kernel<<<(BB * 1024 + T - 1) / T, T, 0, stream>>>(gt_half, p2c, 96, 96, NM, NQ);
    pst_patches_kernel<<<(BB * 256 + T - 1) / T, T, 0, stream>>>(gt_qtr,  p2c, 48, 48, NM, NQ + 1024);

    // 3) pack q = p1+p2 and b = p2c to f16 (K padded 27 -> 32)
    pst_pack_kernel<<<(BB * NM + T - 1) / T, T, 0, stream>>>(p1, p2c, q16, b16);

    // 4) WMMA argmax nearest-neighbor search (unit-norm rows => argmin dist == argmax dot)
    pst_zero_kernel<<<1, 32, 0, stream>>>(acc);
    pst_argmax_wmma_kernel<<<BB * (NQ / 32), 32, 0, stream>>>(q16, b16, ind);

    // 5) gather + L1 mean
    pst_gather_l1_kernel<<<(BB * NQ + T - 1) / T, T, 0, stream>>>(p1, p2c, ind, acc);
    pst_finalize_kernel<<<1, 32, 0, stream>>>(acc, out);
}
